// NERHead_32487132627045
// MI455X (gfx1250) — matrix operand order repaired and checked by reading, not compiled, not run on hardware
//
#include <hip/hip_runtime.h>
#include <hip/hip_bf16.h>
#include <math.h>

// ---------------------------------------------------------------------------
// NER head: emissions = gelu(X @ W1 + b1) @ W2 + b2 ; then CRF -mean LL.
// GEMMs run on v_wmma_f32_16x16x32_bf16 (CDNA5 WMMA, wave32).
// GEMM1: each wave computes a 16x64 strip (4 accumulators) so the f32->bf16
// A-fragment conversion is amortized over 4 WMMAs per K-step.
// ---------------------------------------------------------------------------

typedef __bf16 bf16;
typedef __attribute__((ext_vector_type(16))) __bf16 v16bf;
typedef __attribute__((ext_vector_type(8)))  __bf16 v8bf;
typedef __attribute__((ext_vector_type(8)))  float  v8f;
typedef __attribute__((ext_vector_type(4)))  float  v4f;

#define Bsz 64
#define Lsz 512
#define Dsz 768
#define Hsz 384
#define Tsz 9
#define Msz (Bsz * Lsz)   // 32768 rows

// ---------------------------------------------------------------------------
// Prep: W1 (D x H, f32 row-major) -> W1t (H x D, bf16)   [B-matrix layout]
//       W2 (H x T, f32 row-major) -> W2t (16 x H, bf16)  [T padded to 16]
// ---------------------------------------------------------------------------
__global__ __launch_bounds__(256) void prep_weights(const float* __restrict__ W1,
                                                    const float* __restrict__ W2,
                                                    bf16* __restrict__ W1t,
                                                    bf16* __restrict__ W2t) {
  int i = blockIdx.x * 256 + threadIdx.x;
  if (i < Hsz * Dsz) {                // 294912 entries
    int n = i / Dsz, k = i % Dsz;
    W1t[i] = (bf16)W1[k * Hsz + n];
  } else {
    int j = i - Hsz * Dsz;
    if (j < 16 * Hsz) {               // 6144 entries
      int n = j / Hsz, k = j % Hsz;
      W2t[j] = (bf16)((n < Tsz) ? W2[k * Tsz + n] : 0.0f);
    }
  }
}

// ---------------------------------------------------------------------------
// GEMM1 + GELU: Hout(M x H, bf16) = gelu(X(M x D, f32) @ W1 + b1)
// One 16x64 output strip per wave (4 WMMA tiles); K loop 768 in steps of 32.
// Grid: 2048 M-tiles x 6 N-groups = 12288 waves = 1536 blocks x 8 waves.
// ---------------------------------------------------------------------------
__global__ __launch_bounds__(256) void gemm1_gelu(const float* __restrict__ X,
                                                  const bf16* __restrict__ W1t,
                                                  const float* __restrict__ bias1,
                                                  bf16* __restrict__ Hout) {
  const int wave = threadIdx.x >> 5;
  const int lane = threadIdx.x & 31;
  const int strip = blockIdx.x * 8 + wave;      // 0 .. 12287
  const int m0 = (strip / 6) << 4;              // row block (16 rows)
  const int n0 = (strip % 6) << 6;              // col group (64 cols)
  const int half  = lane >> 4;                  // 0 or 1
  const int l15   = lane & 15;
  const int akoff = half << 3;                  // A: 0 or 8
  const int bkoff = half << 3;                  // B: 0 or 8

  const float* aptr = X + (size_t)(m0 + l15) * Dsz;
  const bf16* bptr0 = W1t + (size_t)(n0 + l15) * Dsz;
  const bf16* bptr1 = bptr0 + (size_t)16 * Dsz;
  const bf16* bptr2 = bptr0 + (size_t)32 * Dsz;
  const bf16* bptr3 = bptr0 + (size_t)48 * Dsz;

  v8f acc0 = {}, acc1 = {}, acc2 = {}, acc3 = {};
  for (int kk = 0; kk < Dsz; kk += 32) {
    // A fragment: rows from f32 global, converted to bf16 in-register.
    v4f a0 = *(const v4f*)(aptr + kk + akoff);
    v4f a1 = *(const v4f*)(aptr + kk + akoff + 4);
    v4f a2 = *(const v4f*)(aptr + kk + akoff + 16);
    v4f a3 = *(const v4f*)(aptr + kk + akoff + 20);
    v16bf a;
#pragma unroll
    for (int i = 0; i < 4; ++i) {
      a[i]      = (bf16)a0[i];
      a[4 + i]  = (bf16)a1[i];
      a[8 + i]  = (bf16)a2[i];
      a[12 + i] = (bf16)a3[i];
    }
    // Four B fragments (8 bf16 at k+8h and 8 at k+16+8h per half) from transposed W1t.
    v8bf p0, p1;
    v16bf b;
#define LOAD_B_AND_MMA(BP, ACC)                                        \
    p0 = *(const v8bf*)((BP) + kk + bkoff);                            \
    p1 = *(const v8bf*)((BP) + kk + bkoff + 16);                       \
    _Pragma("unroll")                                                  \
    for (int i = 0; i < 8; ++i) { b[i] = p0[i]; b[8 + i] = p1[i]; }    \
    ACC = __builtin_amdgcn_wmma_f32_16x16x32_bf16(false, a, false, b,  \
                                                  (short)0, ACC, false, false);
    LOAD_B_AND_MMA(bptr0, acc0)
    LOAD_B_AND_MMA(bptr1, acc1)
    LOAD_B_AND_MMA(bptr2, acc2)
    LOAD_B_AND_MMA(bptr3, acc3)
#undef LOAD_B_AND_MMA
  }

  // Epilogue: bias + exact GELU, store bf16 h.
  v8f accs[4] = {acc0, acc1, acc2, acc3};
#pragma unroll
  for (int j = 0; j < 4; ++j) {
    const int col = n0 + (j << 4) + l15;
    const float bv = bias1[col];
#pragma unroll
    for (int v = 0; v < 8; ++v) {
      const int row = m0 + v + (half << 3);
      float x = accs[j][v] + bv;
      float g = 0.5f * x * (1.0f + erff(x * 0.70710678118654752f));
      Hout[(size_t)row * Hsz + col] = (bf16)g;
    }
  }
}

// ---------------------------------------------------------------------------
// GEMM2: Em(M x 9, f32) = H(M x 384, bf16) @ W2 + b2  (N padded to 16)
// ---------------------------------------------------------------------------
__global__ __launch_bounds__(256) void gemm2_emit(const bf16* __restrict__ Hin,
                                                  const bf16* __restrict__ W2t,
                                                  const float* __restrict__ bias2,
                                                  float* __restrict__ Em) {
  const int wave = threadIdx.x >> 5;
  const int lane = threadIdx.x & 31;
  const int tile = blockIdx.x * 8 + wave;       // 0 .. 2047
  const int m0 = tile << 4;
  const int half  = lane >> 4;
  const int l15   = lane & 15;
  const int akoff = half << 3;
  const int bkoff = half << 3;

  const bf16* aptr = Hin + (size_t)(m0 + l15) * Hsz;
  const bf16* bptr = W2t + (size_t)l15 * Hsz;

  v8f acc = {};
  for (int kk = 0; kk < Hsz; kk += 32) {
    v8bf a0 = *(const v8bf*)(aptr + kk + akoff);
    v8bf a1 = *(const v8bf*)(aptr + kk + akoff + 16);
    v16bf a;
#pragma unroll
    for (int i = 0; i < 8; ++i) { a[i] = a0[i]; a[8 + i] = a1[i]; }
    v8bf b0 = *(const v8bf*)(bptr + kk + bkoff);
    v8bf b1 = *(const v8bf*)(bptr + kk + bkoff + 16);
    v16bf b;
#pragma unroll
    for (int i = 0; i < 8; ++i) { b[i] = b0[i]; b[8 + i] = b1[i]; }
    acc = __builtin_amdgcn_wmma_f32_16x16x32_bf16(false, a, false, b,
                                                  (short)0, acc, false, false);
  }

  if (l15 < Tsz) {
    const float bv = bias2[l15];
#pragma unroll
    for (int v = 0; v < 8; ++v) {
      const int row = m0 + v + (half << 3);
      Em[(size_t)row * Tsz + l15] = acc[v] + bv;
    }
  }
}

// ---------------------------------------------------------------------------
// CRF forward (per batch): one wave per batch. 9 active lanes own one tag
// column each; alpha kept in LDS; numer terms via wave-strided reduction.
// ---------------------------------------------------------------------------
__global__ __launch_bounds__(32) void crf_nll(const float* __restrict__ Em,
                                              const int* __restrict__ tags,
                                              const unsigned char* __restrict__ mask,
                                              const float* __restrict__ start_trans,
                                              const float* __restrict__ end_trans,
                                              const float* __restrict__ trans,
                                              float* __restrict__ per_batch) {
  const int b   = blockIdx.x;
  const int tid = threadIdx.x;
  __shared__ float s_trans[Tsz * Tsz];
  __shared__ float s_alpha[Tsz];
  __shared__ float s_denom;

  const float* E = Em + (size_t)b * Lsz * Tsz;
  const int* tg  = tags + (size_t)b * Lsz;
  const unsigned char* mk = mask + (size_t)b * Lsz;

  for (int i = tid; i < Tsz * Tsz; i += 32) s_trans[i] = trans[i];
  if (tid < Tsz) s_alpha[tid] = start_trans[tid] + E[tid];
  __syncthreads();

  float trcol[Tsz];
  if (tid < Tsz) {
#pragma unroll
    for (int i = 0; i < Tsz; ++i) trcol[i] = s_trans[i * Tsz + tid];
  }

  for (int t = 1; t < Lsz; ++t) {
    float a[Tsz];
    if (tid < Tsz) {
#pragma unroll
      for (int i = 0; i < Tsz; ++i) a[i] = s_alpha[i];
    }
    __syncthreads();
    if (tid < Tsz) {
      float mx = -3.0e38f;
#pragma unroll
      for (int i = 0; i < Tsz; ++i) mx = fmaxf(mx, a[i] + trcol[i]);
      float s = 0.0f;
#pragma unroll
      for (int i = 0; i < Tsz; ++i) s += __expf(a[i] + trcol[i] - mx);
      float nxt = mx + __logf(s) + E[t * Tsz + tid];
      if (mk[t]) s_alpha[tid] = nxt;
    }
    __syncthreads();
  }

  // denom = logsumexp(alpha + end_trans)
  float dv = (tid < Tsz) ? (s_alpha[tid] + end_trans[tid]) : -3.0e38f;
  float mx = dv;
  for (int off = 16; off > 0; off >>= 1) mx = fmaxf(mx, __shfl_down(mx, off, 32));
  mx = __shfl(mx, 0, 32);
  float ex = (tid < Tsz) ? __expf(dv - mx) : 0.0f;
  for (int off = 16; off > 0; off >>= 1) ex += __shfl_down(ex, off, 32);
  if (tid == 0) s_denom = mx + __logf(ex);

  // numer: emit + transition scores + start/end
  float emit = 0.0f, trs = 0.0f;
  int cnt = 0;
  for (int t = tid; t < Lsz; t += 32) {
    const float m = (float)mk[t];
    emit += E[t * Tsz + tg[t]] * m;
    cnt  += mk[t] ? 1 : 0;
    if (t >= 1) trs += s_trans[tg[t - 1] * Tsz + tg[t]] * m;
  }
  for (int off = 16; off > 0; off >>= 1) {
    emit += __shfl_down(emit, off, 32);
    trs  += __shfl_down(trs, off, 32);
    cnt  += __shfl_down(cnt, off, 32);
  }
  __syncthreads();
  if (tid == 0) {
    const int last_idx = cnt - 1;
    const int last_tag = tg[last_idx];
    const float numer = start_trans[tg[0]] + emit + trs + end_trans[last_tag];
    per_batch[b] = numer - s_denom;
  }
}

__global__ void finalize_mean(const float* __restrict__ per_batch,
                              float* __restrict__ out) {
  if (threadIdx.x == 0 && blockIdx.x == 0) {
    float s = 0.0f;
    for (int i = 0; i < Bsz; ++i) s += per_batch[i];
    out[0] = -s / (float)Bsz;
  }
}

// ---------------------------------------------------------------------------
extern "C" void kernel_launch(void* const* d_in, const int* in_sizes, int n_in,
                              void* d_out, int out_size, void* d_ws, size_t ws_size,
                              hipStream_t stream) {
  const float* X           = (const float*)d_in[0];          // (B,L,D)
  const int*   tags        = (const int*)d_in[1];            // (B,L)
  const unsigned char* msk = (const unsigned char*)d_in[2];  // (B,L) bool
  const float* W1          = (const float*)d_in[3];          // (D,H)
  const float* b1          = (const float*)d_in[4];          // (H)
  const float* W2          = (const float*)d_in[5];          // (H,T)
  const float* b2          = (const float*)d_in[6];          // (T)
  const float* start_trans = (const float*)d_in[7];          // (T)
  const float* end_trans   = (const float*)d_in[8];          // (T)
  const float* trans       = (const float*)d_in[9];          // (T,T)

  // Workspace carve-out (all 256B-aligned offsets).
  char* ws = (char*)d_ws;
  bf16*  W1t = (bf16*)(ws + 0);                         //   589,824 B
  bf16*  W2t = (bf16*)(ws + 589824);                    //    12,288 B
  bf16*  Hb  = (bf16*)(ws + 602112);                    // 25,165,824 B
  float* Em  = (float*)(ws + 25767936);                 //  1,179,648 B
  float* pb  = (float*)(ws + 26947584);                 //       256 B

  // 1) transpose/convert weights to bf16 [N][K]
  prep_weights<<<(Hsz * Dsz + 16 * Hsz + 255) / 256, 256, 0, stream>>>(W1, W2, W1t, W2t);

  // 2) h = gelu(X @ W1 + b1)   : 2048 M-tiles x 6 N-groups, 8 waves/block
  gemm1_gelu<<<(2048 * 6) / 8, 256, 0, stream>>>(X, W1t, b1, Hb);

  // 3) emissions = h @ W2 + b2 : 2048 tiles, 8 waves/block
  gemm2_emit<<<2048 / 8, 256, 0, stream>>>(Hb, W2t, b2, Em);

  // 4) CRF NLL per batch (one wave per batch)
  crf_nll<<<Bsz, 32, 0, stream>>>(Em, tags, msk, start_trans, end_trans, trans, pb);

  // 5) -mean
  finalize_mean<<<1, 32, 0, stream>>>(pb, (float*)d_out);
}
